// GoatLayerIMP4_87720412054001
// MI455X (gfx1250) — matrix operand order repaired and checked by reading, not compiled, not run on hardware
//
#include <hip/hip_runtime.h>
#include <hip/hip_bf16.h>

// ---------------------------------------------------------------------------
// GOAT layer (GAT attention + per-node LSTM aggregation) for MI455X / gfx1250.
// f16 WMMA (v_wmma_f32_16x16x32_f16) with f32 accumulate for all GEMMs; LSTM
// recurrence fused per 16-node tile so gates never touch HBM. Gate activations
// use the CDNA5 hardware v_tanh_f32 TRANS op (sigmoid via 0.5*tanh(x/2)+0.5).
// ---------------------------------------------------------------------------

typedef __attribute__((ext_vector_type(16))) _Float16 v16h;
typedef __attribute__((ext_vector_type(8)))  _Float16 h8;
typedef __attribute__((ext_vector_type(8)))  float    v8f;

#define FIN_  256
#define FOUT_ 128
#define H_    128
#define DEG_  16
#define XTLD  136   // padded f16 row stride (272B = 17 banks*4) -> conflict-free A reads

__device__ __forceinline__ v16h pack16(h8 lo, h8 hi) {
    v16h r;
#pragma unroll
    for (int i = 0; i < 8; i++) { r[i] = lo[i]; r[8 + i] = hi[i]; }
    return r;
}

// A fragment (16x32 f16, ISA 7.12.2): lane m = lane&15, half = lane>>4.
// VGPR0..3 hold K = half*8 + 0..7 (contiguous), VGPR4..7 hold K = 16 + half*8 + 0..7.
__device__ __forceinline__ v16h load_a16(const _Float16* src, int ld, int k0, int lane) {
    const int half = lane >> 4;
    const int m    = lane & 15;
    const _Float16* p = src + (size_t)m * ld + k0 + half * 8;
    h8 lo = *(const h8*)(p);
    h8 hi = *(const h8*)(p + 16);
    return pack16(lo, hi);
}

// B fragment (32x16 f16, K x N) from row-major W[n][k] (ldk = K stride):
// lane n = lane&15, half = lane>>4; VGPR0..3 hold K = half*8 + 0..7,
// VGPR4..7 hold K = 16 + half*8 + 0..7 (same order as A).
__device__ __forceinline__ v16h load_b16(const _Float16* W, int ldk, int n, int k0, int lane) {
    const int half = lane >> 4;
    const _Float16* p = W + (size_t)n * ldk + k0 + half * 8;
    h8 lo = *(const h8*)(p);
    h8 hi = *(const h8*)(p + 16);
    return pack16(lo, hi);
}

// CDNA5 hardware tanh (TRANS op, co-executes with WMMA); exact-identity sigmoid.
#if defined(__has_builtin) && __has_builtin(__builtin_amdgcn_tanhf)
__device__ __forceinline__ float fast_tanh(float x) { return __builtin_amdgcn_tanhf(x); }
#else
__device__ __forceinline__ float fast_tanh(float x) { return tanhf(x); }
#endif
__device__ __forceinline__ float fast_sigmoid(float x) { return fmaf(0.5f, fast_tanh(0.5f * x), 0.5f); }

// ---------------------------------------------------------------------------
// Small utility kernels
// ---------------------------------------------------------------------------
__global__ void cvt_f32_to_f16(const float* __restrict__ in, _Float16* __restrict__ out, int n) {
    for (int i = blockIdx.x * blockDim.x + threadIdx.x; i < n; i += gridDim.x * blockDim.x)
        out[i] = (_Float16)in[i];
}

__global__ void bias_sum_kernel(const float* __restrict__ a, const float* __restrict__ b,
                                float* __restrict__ o, int n) {
    int i = blockIdx.x * blockDim.x + threadIdx.x;
    if (i < n) o[i] = a[i] + b[i];
}

// proj = x @ W_proj^T : block = 16 rows x 128 cols, 8 waves, 1 tile each, K=256
__global__ __launch_bounds__(256) void proj_gemm_kernel(const _Float16* __restrict__ xh,
                                                        const _Float16* __restrict__ wproj,
                                                        float* __restrict__ proj) {
    const int t = threadIdx.x, lane = t & 31, wave = t >> 5;
    const int half = lane >> 4, n16 = lane & 15;
    const int base = blockIdx.x * 16;
    const int ncol = wave * 16 + n16;
    const _Float16* arow = xh + (size_t)base * FIN_;
    v8f acc;
#pragma unroll
    for (int v = 0; v < 8; v++) acc[v] = 0.0f;
#pragma unroll
    for (int kc = 0; kc < 8; kc++) {
        v16h a = load_a16(arow, FIN_, kc * 32, lane);
        v16h b = load_b16(wproj, FIN_, ncol, kc * 32, lane);
        acc = __builtin_amdgcn_wmma_f32_16x16x32_f16(false, a, false, b, (short)0, acc, false, false);
    }
#pragma unroll
    for (int v = 0; v < 8; v++) {
        int m = v + half * 8;
        proj[(size_t)(base + m) * FOUT_ + ncol] = acc[v];
    }
}

// s_src[i] = proj[i]·a_src ; s_trg[i] = proj[i]·a_trg  (one wave per node)
__global__ __launch_bounds__(256) void attn_scalar_kernel(const float* __restrict__ proj,
                                                          const float* __restrict__ a_src,
                                                          const float* __restrict__ a_trg,
                                                          float* __restrict__ s_src,
                                                          float* __restrict__ s_trg, int N) {
    int warp = (blockIdx.x * blockDim.x + threadIdx.x) >> 5;
    int lane = threadIdx.x & 31;
    if (warp >= N) return;
    const float* p = proj + (size_t)warp * FOUT_;
    float a1 = 0.f, a2 = 0.f;
#pragma unroll
    for (int c = 0; c < FOUT_ / 32; c++) {
        float v = p[lane + c * 32];
        a1 += v * a_src[lane + c * 32];
        a2 += v * a_trg[lane + c * 32];
    }
#pragma unroll
    for (int off = 16; off > 0; off >>= 1) {
        a1 += __shfl_down(a1, off, 32);
        a2 += __shfl_down(a2, off, 32);
    }
    if (lane == 0) { s_src[warp] = a1; s_trg[warp] = a2; }
}

// two-pass global max of leaky_relu(s_src[src]+s_trg[trg], 0.2)
__global__ __launch_bounds__(256) void score_max_part_kernel(const float* __restrict__ s_src,
                                                             const float* __restrict__ s_trg,
                                                             const int* __restrict__ src,
                                                             const int* __restrict__ trg,
                                                             int E, float* __restrict__ part) {
    __shared__ float red[256];
    float m = -3.4e38f;
    for (int e = blockIdx.x * blockDim.x + threadIdx.x; e < E; e += gridDim.x * blockDim.x) {
        float sc = s_src[src[e]] + s_trg[trg[e]];
        sc = sc > 0.f ? sc : 0.2f * sc;
        m = fmaxf(m, sc);
    }
    red[threadIdx.x] = m;
    __syncthreads();
    for (int s = 128; s > 0; s >>= 1) {
        if (threadIdx.x < s) red[threadIdx.x] = fmaxf(red[threadIdx.x], red[threadIdx.x + s]);
        __syncthreads();
    }
    if (threadIdx.x == 0) part[blockIdx.x] = red[0];
}

__global__ __launch_bounds__(256) void score_max_final_kernel(const float* __restrict__ part,
                                                              int n, float* __restrict__ gmax) {
    __shared__ float red[256];
    float m = -3.4e38f;
    for (int i = threadIdx.x; i < n; i += 256) m = fmaxf(m, part[i]);
    red[threadIdx.x] = m;
    __syncthreads();
    for (int s = 128; s > 0; s >>= 1) {
        if (threadIdx.x < s) red[threadIdx.x] = fmaxf(red[threadIdx.x], red[threadIdx.x + s]);
        __syncthreads();
    }
    if (threadIdx.x == 0) gmax[0] = red[0];
}

// ---------------------------------------------------------------------------
// Fused: softmax(16 edges) -> sort -> 16-step LSTM (WMMA) -> skip GEMM -> out
// One block = 16 nodes. 8 waves. LDS ~60.5 KB.
// ---------------------------------------------------------------------------
__global__ __launch_bounds__(256) void goat_fused_kernel(
    const float* __restrict__ proj, const float* __restrict__ s_src,
    const float* __restrict__ s_trg, const int* __restrict__ src_idx,
    const float* __restrict__ gmax_p, const _Float16* __restrict__ xh,
    const _Float16* __restrict__ wih, const _Float16* __restrict__ whh,
    const float* __restrict__ bsum, const _Float16* __restrict__ wskip,
    const float* __restrict__ bias, float* __restrict__ out) {
    __shared__ float    att_s[16][16];
    __shared__ int      srcs[16][16];
    __shared__ float    seq_att[16][16];
    __shared__ int      seq_src[16][16];
    __shared__ _Float16 xt16[16 * XTLD];   // x_t tile (f16, padded)
    __shared__ _Float16 h16[16 * XTLD];    // h state as WMMA A operand
    __shared__ float    h32[16 * H_];      // h state f32
    __shared__ float    c32[16 * H_];      // c state f32
    __shared__ float    gbuf[16 * 4 * H_]; // gates [16][512]

    const int t = threadIdx.x, lane = t & 31, wave = t >> 5;
    const int half = lane >> 4, n16 = lane & 15;
    const int base = blockIdx.x * 16;
    const float gmax = gmax_p[0];

    // ---- phase 1: edge scores (each thread one of 16x16 edges) ----
    {
        const int il = t >> 4, j = t & 15;
        const int node = base + il;
        const int e = node * DEG_ + j;
        const int s = src_idx[e];
        float sc = s_src[s] + s_trg[node];
        sc = sc > 0.f ? sc : 0.2f * sc;
        att_s[il][j] = __expf(sc - gmax);
        srcs[il][j] = s;
    }
    for (int i = t; i < 16 * H_; i += 256) { h32[i] = 0.f; c32[i] = 0.f; }
    for (int i = t; i < 16 * XTLD; i += 256) h16[i] = (_Float16)0.f;
    __syncthreads();

    // ---- phase 2: per-node softmax + stable ascending sort, emit reversed ----
    if (t < 16) {
        float d = 1e-16f;
#pragma unroll
        for (int k = 0; k < 16; k++) d += att_s[t][k];
        float av[16]; int sv[16];
#pragma unroll
        for (int k = 0; k < 16; k++) {
            float key = att_s[t][k] / d;
            int   si  = srcs[t][k];
            int p = k;
            while (p > 0 && av[p - 1] > key) { av[p] = av[p - 1]; sv[p] = sv[p - 1]; p--; }
            av[p] = key; sv[p] = si;
        }
        // reversed ascending-stable == seq[:, ::-1] order of the reference
#pragma unroll
        for (int k = 0; k < 16; k++) { seq_att[t][k] = av[15 - k]; seq_src[t][k] = sv[15 - k]; }
    }
    __syncthreads();

    // ---- phase 3: 16-step LSTM, gates = x_t@w_ih^T + h@w_hh^T + b ----
    for (int step = 0; step < DEG_; step++) {
        {   // build x_t tile: 16 threads/row, 8 cols each
            const int m = t >> 4, c0 = (t & 15) * 8;
            const int   sidx = seq_src[m][step];
            const float a    = seq_att[m][step];
            const float* pr  = proj + (size_t)sidx * FOUT_ + c0;
#pragma unroll
            for (int q = 0; q < 8; q++) xt16[m * XTLD + c0 + q] = (_Float16)(pr[q] * a);
        }
        __syncthreads();
#pragma unroll
        for (int tile = 0; tile < 4; tile++) {
            const int ncol = wave * 64 + tile * 16 + n16;  // 0..511 gate column
            v8f acc;
            const float bv = bsum[ncol];
#pragma unroll
            for (int v = 0; v < 8; v++) acc[v] = bv;
#pragma unroll
            for (int kc = 0; kc < 4; kc++) {
                v16h ax = load_a16(xt16, XTLD, kc * 32, lane);
                v16h bx = load_b16(wih, FOUT_, ncol, kc * 32, lane);
                acc = __builtin_amdgcn_wmma_f32_16x16x32_f16(false, ax, false, bx, (short)0, acc, false, false);
                v16h ah = load_a16(h16, XTLD, kc * 32, lane);
                v16h bh = load_b16(whh, H_, ncol, kc * 32, lane);
                acc = __builtin_amdgcn_wmma_f32_16x16x32_f16(false, ah, false, bh, (short)0, acc, false, false);
            }
#pragma unroll
            for (int v = 0; v < 8; v++) {
                int m = v + half * 8;
                gbuf[m * (4 * H_) + ncol] = acc[v];
            }
        }
        __syncthreads();
        // elementwise LSTM cell update: 2048 (m,j) pairs over 256 threads.
        // 5 transcendentals/elem -> single-instruction v_tanh_f32 each.
#pragma unroll
        for (int q = 0; q < 8; q++) {
            const int idx = t + q * 256;
            const int m = idx >> 7, j = idx & (H_ - 1);
            const float ig = gbuf[m * (4 * H_) + j];
            const float fg = gbuf[m * (4 * H_) + H_ + j];
            const float gg = gbuf[m * (4 * H_) + 2 * H_ + j];
            const float og = gbuf[m * (4 * H_) + 3 * H_ + j];
            float c = fast_sigmoid(fg) * c32[idx] + fast_sigmoid(ig) * fast_tanh(gg);
            float h = fast_sigmoid(og) * fast_tanh(c);
            c32[idx] = c;
            h32[idx] = h;
            h16[m * XTLD + j] = (_Float16)h;
        }
        __syncthreads();
    }

    // ---- phase 4: out = lrelu(h + x@W_skip^T + bias, 0.01) ----
    {
        const int ncol = wave * 16 + n16;  // 0..127
        const _Float16* arow = xh + (size_t)base * FIN_;
        v8f acc;
        const float bv = bias[ncol];
#pragma unroll
        for (int v = 0; v < 8; v++) acc[v] = bv;
#pragma unroll
        for (int kc = 0; kc < 8; kc++) {
            v16h a = load_a16(arow, FIN_, kc * 32, lane);
            v16h b = load_b16(wskip, FIN_, ncol, kc * 32, lane);
            acc = __builtin_amdgcn_wmma_f32_16x16x32_f16(false, a, false, b, (short)0, acc, false, false);
        }
#pragma unroll
        for (int v = 0; v < 8; v++) {
            const int m = v + half * 8;
            float o = acc[v] + h32[m * H_ + ncol];
            o = o > 0.f ? o : 0.01f * o;
            out[(size_t)(base + m) * FOUT_ + ncol] = o;
        }
    }
}

// ---------------------------------------------------------------------------
extern "C" void kernel_launch(void* const* d_in, const int* in_sizes, int n_in,
                              void* d_out, int out_size, void* d_ws, size_t ws_size,
                              hipStream_t stream) {
    const float* x      = (const float*)d_in[0];
    const float* Wproj  = (const float*)d_in[1];
    const float* a_src  = (const float*)d_in[2];
    const float* a_trg  = (const float*)d_in[3];
    const float* Wskip  = (const float*)d_in[4];
    const float* bias   = (const float*)d_in[5];
    const float* w_ih_b = (const float*)d_in[10];
    const float* w_hh_b = (const float*)d_in[11];
    const float* b_ih_b = (const float*)d_in[12];
    const float* b_hh_b = (const float*)d_in[13];
    const int*   ei     = (const int*)d_in[14];

    const int N = in_sizes[0] / FIN_;
    const int E = in_sizes[14] / 2;
    const int* src = ei;
    const int* trg = ei + E;

    // workspace carve (~52 MB)
    char* w = (char*)d_ws;
    size_t off = 0;
    auto carve = [&](size_t bytes) -> void* {
        void* p = w + off;
        off = (off + bytes + 255) & ~(size_t)255;
        return p;
    };
    _Float16* xh      = (_Float16*)carve((size_t)N * FIN_ * sizeof(_Float16));
    _Float16* wproj_h = (_Float16*)carve((size_t)FOUT_ * FIN_ * sizeof(_Float16));
    _Float16* wskip_h = (_Float16*)carve((size_t)FOUT_ * FIN_ * sizeof(_Float16));
    _Float16* wih_h   = (_Float16*)carve((size_t)4 * H_ * FOUT_ * sizeof(_Float16));
    _Float16* whh_h   = (_Float16*)carve((size_t)4 * H_ * H_ * sizeof(_Float16));
    float*    bsum    = (float*)carve((size_t)4 * H_ * sizeof(float));
    float*    proj    = (float*)carve((size_t)N * FOUT_ * sizeof(float));
    float*    s_src   = (float*)carve((size_t)N * sizeof(float));
    float*    s_trg   = (float*)carve((size_t)N * sizeof(float));
    float*    partmax = (float*)carve(1024 * sizeof(float));
    float*    gmax    = (float*)carve(256);

    // 1) precision conversion
    cvt_f32_to_f16<<<1024, 256, 0, stream>>>(x, xh, N * FIN_);
    cvt_f32_to_f16<<<64, 256, 0, stream>>>(Wproj, wproj_h, FOUT_ * FIN_);
    cvt_f32_to_f16<<<64, 256, 0, stream>>>(Wskip, wskip_h, FOUT_ * FIN_);
    cvt_f32_to_f16<<<64, 256, 0, stream>>>(w_ih_b, wih_h, 4 * H_ * FOUT_);
    cvt_f32_to_f16<<<64, 256, 0, stream>>>(w_hh_b, whh_h, 4 * H_ * H_);
    bias_sum_kernel<<<2, 256, 0, stream>>>(b_ih_b, b_hh_b, bsum, 4 * H_);

    // 2) proj GEMM (WMMA)
    proj_gemm_kernel<<<N / 16, 256, 0, stream>>>(xh, wproj_h, proj);

    // 3) attention scalars
    attn_scalar_kernel<<<(N * 32 + 255) / 256, 256, 0, stream>>>(proj, a_src, a_trg, s_src, s_trg, N);

    // 4) global score max (deterministic two-pass)
    score_max_part_kernel<<<512, 256, 0, stream>>>(s_src, s_trg, src, trg, E, partmax);
    score_max_final_kernel<<<1, 256, 0, stream>>>(partmax, 512, gmax);

    // 5) fused softmax + sort + LSTM + skip
    goat_fused_kernel<<<N / 16, 256, 0, stream>>>(proj, s_src, s_trg, src, gmax, xh,
                                                  wih_h, whh_h, bsum, wskip_h, bias,
                                                  (float*)d_out);
}
